// Qwen2MoeMLP_9440338117223
// MI455X (gfx1250) — matrix operand order repaired and checked by reading, not compiled, not run on hardware
//
#include <hip/hip_runtime.h>
#include <hip/hip_bf16.h>

typedef __attribute__((ext_vector_type(16))) __bf16 v16bf;
typedef __attribute__((ext_vector_type(8)))  __bf16 v8bf;
typedef __attribute__((ext_vector_type(8)))  float  v8f;

#define T_TOK 1024
#define HDIM  2048
#define IDIM  1408
#define NEXP  16
#define TOPK  4
#define MTILE 32                    // tokens per workgroup (2 WMMA M-tiles)
#define IPAD  (IDIM + 8)

__device__ __forceinline__ float sigmoid_fast(float v) {
    return 1.0f / (1.0f + __expf(-v));
}

union ABfrag { v16bf v; v8bf h[2]; };

// -------- Kernel 0: bulk fp32 -> bf16 conversion (weights, activations) ---
__global__ void cvt_f32_bf16(const float* __restrict__ src,
                             __bf16* __restrict__ dst, int n8) {
    int i = blockIdx.x * blockDim.x + threadIdx.x;
    if (i >= n8) return;
    const float* s = src + (size_t)i * 8;
    float4 f0 = *(const float4*)(s);
    float4 f1 = *(const float4*)(s + 4);
    v8bf o;
    o[0] = (__bf16)f0.x; o[1] = (__bf16)f0.y; o[2] = (__bf16)f0.z; o[3] = (__bf16)f0.w;
    o[4] = (__bf16)f1.x; o[5] = (__bf16)f1.y; o[6] = (__bf16)f1.z; o[7] = (__bf16)f1.w;
    *(v8bf*)(dst + (size_t)i * 8) = o;
}

// -------- Kernel 1: per-token softmax + top-4 + renorm --------------------
__global__ void route_topk(const float* __restrict__ logits,
                           int* __restrict__ tidx, float* __restrict__ tw) {
    int t = blockIdx.x * blockDim.x + threadIdx.x;
    if (t >= T_TOK) return;
    float p[NEXP];
    float mx = -3.4e38f;
#pragma unroll
    for (int e = 0; e < NEXP; ++e) { p[e] = logits[t * NEXP + e]; mx = fmaxf(mx, p[e]); }
    float sum = 0.f;
#pragma unroll
    for (int e = 0; e < NEXP; ++e) { p[e] = __expf(p[e] - mx); sum += p[e]; }
    float inv = 1.f / sum;
#pragma unroll
    for (int e = 0; e < NEXP; ++e) p[e] *= inv;
    int   bi[TOPK];
    float bv[TOPK];
    float wsum = 0.f;
#pragma unroll
    for (int s = 0; s < TOPK; ++s) {
        int best = 0; float v = -1.f;
#pragma unroll
        for (int e = 0; e < NEXP; ++e)
            if (p[e] > v) { v = p[e]; best = e; }   // ties -> lowest index
        bi[s] = best; bv[s] = v; wsum += v; p[best] = -2.f;
    }
    float wn = 1.f / wsum;
#pragma unroll
    for (int s = 0; s < TOPK; ++s) {
        tidx[t * TOPK + s] = bi[s];
        tw[t * TOPK + s]   = bv[s] * wn;
    }
}

// -------- Kernel 2: stable per-expert token-list build (deterministic) ----
__global__ void build_lists(const int* __restrict__ tidx,
                            int* __restrict__ count, int* __restrict__ list) {
    __shared__ int scan[T_TOK];
    const int e = blockIdx.x;
    const int t = threadIdx.x;
    int slot = -1;
#pragma unroll
    for (int s = 0; s < TOPK; ++s)
        if (tidx[t * TOPK + s] == e) slot = s;
    int flag = (slot >= 0) ? 1 : 0;
    scan[t] = flag;
    __syncthreads();
    for (int off = 1; off < T_TOK; off <<= 1) {
        int v = scan[t];
        if (t >= off) v += scan[t - off];
        __syncthreads();
        scan[t] = v;
        __syncthreads();
    }
    if (flag) list[e * T_TOK + (scan[t] - 1)] = (t << 2) | slot;
    if (t == T_TOK - 1) count[e] = scan[t];
}

// -------- Kernel 3: per-(expert, 32-token tile) SwiGLU via bf16 WMMA ------
__global__ __launch_bounds__(256, 1)
void moe_expert_tile(const __bf16* __restrict__ xb,
                     const __bf16* __restrict__ gwb,
                     const __bf16* __restrict__ uwb,
                     const __bf16* __restrict__ dwb,
                     const int*    __restrict__ count,
                     const int*    __restrict__ list,
                     const float*  __restrict__ tw,
                     float*        __restrict__ partial) {
    const int e    = blockIdx.y;
    const int tile = blockIdx.x;
    const int cnt  = count[e];
    if (tile * MTILE >= cnt) return;

    __shared__ __bf16 hs[MTILE][IPAD];   // silu(g)*u, 32 x 1408 bf16 (~90 KB)
    __shared__ int    s_tok[MTILE];
    __shared__ int    s_prow[MTILE];     // token*4+slot, or -1 for pad lanes
    __shared__ float  s_w[MTILE];

    const int tid = threadIdx.x;
    if (tid < MTILE) {
        int idx   = tile * MTILE + tid;
        int valid = idx < cnt;
        int ta    = valid ? list[e * T_TOK + idx] : 0;
        int tok   = ta >> 2;
        int slot  = ta & 3;
        s_tok[tid]  = valid ? tok : 0;
        s_prow[tid] = valid ? (tok * TOPK + slot) : -1;
        s_w[tid]    = valid ? tw[tok * TOPK + slot] : 0.f;
    }
    __syncthreads();

    const int lane = tid & 31;
    const int wave = tid >> 5;
    const int g    = lane >> 4;     // half-wave id (K split)
    const int mn   = lane & 15;     // A row (M) / B column (N) within a tile

    const __bf16* __restrict__ gwE = gwb + (size_t)e * IDIM * HDIM;
    const __bf16* __restrict__ uwE = uwb + (size_t)e * IDIM * HDIM;
    const __bf16* __restrict__ dwE = dwb + (size_t)e * HDIM * IDIM;
    const __bf16* __restrict__ xr0 = xb + (size_t)s_tok[mn]      * HDIM;
    const __bf16* __restrict__ xr1 = xb + (size_t)s_tok[16 + mn] * HDIM;

    // ---- Stage A: G,U = X(32xH) * W^T over 88 N-tiles; fused SiLU -> hs ----
    // Wave handles N-tiles j = 0..10 (tile = wave + 8*j) in groups of 2.
    for (int jg = 0; jg < 6; ++jg) {
        v8f accg[2][2] = {};
        v8f accu[2][2] = {};
        int  ntile[2];
        bool vld[2];
#pragma unroll
        for (int q = 0; q < 2; ++q) {
            int j    = jg * 2 + q;
            vld[q]   = (j < 11);
            ntile[q] = vld[q] ? (wave + 8 * j) : wave;
        }
        for (int k0 = 0; k0 < HDIM; k0 += 32) {
            ABfrag a0, a1;
            a0.h[0] = *(const v8bf*)(xr0 + k0 + g * 16);
            a0.h[1] = *(const v8bf*)(xr0 + k0 + g * 16 + 8);
            a1.h[0] = *(const v8bf*)(xr1 + k0 + g * 16);
            a1.h[1] = *(const v8bf*)(xr1 + k0 + g * 16 + 8);
#pragma unroll
            for (int q = 0; q < 2; ++q) {
                const int ncol = ntile[q] * 16 + mn;
                const __bf16* bgp = gwE + (size_t)ncol * HDIM + k0 + g * 16;
                const __bf16* bup = uwE + (size_t)ncol * HDIM + k0 + g * 16;
                __builtin_prefetch(bgp + 32, 0, 0);
                __builtin_prefetch(bup + 32, 0, 0);
                v16bf bg = *(const v16bf*)bgp;
                v16bf bu = *(const v16bf*)bup;
                accg[0][q] = __builtin_amdgcn_wmma_f32_16x16x32_bf16(
                    false, a0.v, false, bg, (short)0, accg[0][q], false, false);
                accg[1][q] = __builtin_amdgcn_wmma_f32_16x16x32_bf16(
                    false, a1.v, false, bg, (short)0, accg[1][q], false, false);
                accu[0][q] = __builtin_amdgcn_wmma_f32_16x16x32_bf16(
                    false, a0.v, false, bu, (short)0, accu[0][q], false, false);
                accu[1][q] = __builtin_amdgcn_wmma_f32_16x16x32_bf16(
                    false, a1.v, false, bu, (short)0, accu[1][q], false, false);
            }
        }
#pragma unroll
        for (int q = 0; q < 2; ++q) {
            if (!vld[q]) continue;
#pragma unroll
            for (int mt = 0; mt < 2; ++mt) {
#pragma unroll
                for (int r = 0; r < 8; ++r) {
                    int   m  = mt * 16 + g * 8 + r;   // C layout: M = r + 8*(lane/16)
                    float gv = accg[mt][q][r];
                    float uv = accu[mt][q][r];
                    float h  = gv * sigmoid_fast(gv) * uv;
                    hs[m][ntile[q] * 16 + mn] = (__bf16)h;
                }
            }
        }
    }
    __syncthreads();

    // ---- Stage C: P = hs(32xI) * down^T over 128 N-tiles; scale + scatter --
    for (int jg = 0; jg < 8; ++jg) {
        v8f acc[2][2] = {};
        int ncolq[2];
#pragma unroll
        for (int q = 0; q < 2; ++q)
            ncolq[q] = (wave + 8 * (jg * 2 + q)) * 16 + mn;   // 128 = 8 waves * 16
        for (int k0 = 0; k0 < IDIM; k0 += 32) {
            ABfrag a0, a1;
            a0.h[0] = *(const v8bf*)(&hs[mn][k0 + g * 16]);
            a0.h[1] = *(const v8bf*)(&hs[mn][k0 + g * 16 + 8]);
            a1.h[0] = *(const v8bf*)(&hs[16 + mn][k0 + g * 16]);
            a1.h[1] = *(const v8bf*)(&hs[16 + mn][k0 + g * 16 + 8]);
#pragma unroll
            for (int q = 0; q < 2; ++q) {
                const __bf16* bp = dwE + (size_t)ncolq[q] * IDIM + k0 + g * 16;
                __builtin_prefetch(bp + 32, 0, 0);
                v16bf b = *(const v16bf*)bp;
                acc[0][q] = __builtin_amdgcn_wmma_f32_16x16x32_bf16(
                    false, a0.v, false, b, (short)0, acc[0][q], false, false);
                acc[1][q] = __builtin_amdgcn_wmma_f32_16x16x32_bf16(
                    false, a1.v, false, b, (short)0, acc[1][q], false, false);
            }
        }
#pragma unroll
        for (int q = 0; q < 2; ++q) {
#pragma unroll
            for (int mt = 0; mt < 2; ++mt) {
#pragma unroll
                for (int r = 0; r < 8; ++r) {
                    int m    = mt * 16 + g * 8 + r;
                    int prow = s_prow[m];
                    if (prow >= 0)
                        partial[(size_t)prow * HDIM + ncolq[q]] = acc[mt][q][r] * s_w[m];
                }
            }
        }
    }
}

// -------- Kernel 4: deterministic combine of the 4 slot partials ----------
__global__ void combine_topk(const float* __restrict__ partial,
                             float* __restrict__ out) {
    int i = blockIdx.x * blockDim.x + threadIdx.x;   // over T*H/4 float4s
    if (i >= T_TOK * HDIM / 4) return;
    int n = i * 4;
    int t = n >> 11;         // / HDIM
    int h = n & (HDIM - 1);  // % HDIM
    const float* base = partial + (size_t)t * TOPK * HDIM + h;
    float4 s = *(const float4*)(base);
    float4 a = *(const float4*)(base + (size_t)1 * HDIM);
    float4 b = *(const float4*)(base + (size_t)2 * HDIM);
    float4 c = *(const float4*)(base + (size_t)3 * HDIM);
    s.x += a.x + b.x + c.x;
    s.y += a.y + b.y + c.y;
    s.z += a.z + b.z + c.z;
    s.w += a.w + b.w + c.w;
    ((float4*)out)[i] = s;
}

extern "C" void kernel_launch(void* const* d_in, const int* in_sizes, int n_in,
                              void* d_out, int out_size, void* d_ws, size_t ws_size,
                              hipStream_t stream) {
    (void)in_sizes; (void)n_in; (void)out_size; (void)ws_size;
    const float* x      = (const float*)d_in[0];
    const float* gw     = (const float*)d_in[1];
    const float* uw     = (const float*)d_in[2];
    const float* dw     = (const float*)d_in[3];
    const float* logits = (const float*)d_in[4];
    // d_in[5] = top_k (fixed at 4 per setup)

    const size_t WELEM = (size_t)NEXP * IDIM * HDIM;   // 46,137,344 per tensor
    const size_t XELEM = (size_t)T_TOK * HDIM;         // 2,097,152

    char*   ws      = (char*)d_ws;
    int*    tidx    = (int*)   (ws + 0);          // 16 KB
    float*  tw      = (float*) (ws + 16384);      // 16 KB
    int*    cnt     = (int*)   (ws + 32768);      // 64 B
    int*    lst     = (int*)   (ws + 36864);      // 64 KB
    float*  partial = (float*) (ws + 131072);     // 32 MB
    __bf16* xbb     = (__bf16*)(ws + 33685504);   // 4 MB
    __bf16* gwb     = (__bf16*)(ws + 37879808);   // 88 MB
    __bf16* uwb     = (__bf16*)(ws + 130154496);  // 88 MB
    __bf16* dwb     = (__bf16*)(ws + 222429184);  // 88 MB  (total ~300 MB)

    // One-time-per-call precision downconversion (bandwidth-amortized:
    // every expert's weights are re-read by ~8 tiles afterwards).
    cvt_f32_bf16<<<(int)(WELEM / 8 / 256), 256, 0, stream>>>(gw, gwb, (int)(WELEM / 8));
    cvt_f32_bf16<<<(int)(WELEM / 8 / 256), 256, 0, stream>>>(uw, uwb, (int)(WELEM / 8));
    cvt_f32_bf16<<<(int)(WELEM / 8 / 256), 256, 0, stream>>>(dw, dwb, (int)(WELEM / 8));
    cvt_f32_bf16<<<(int)(XELEM / 8 / 256), 256, 0, stream>>>(x, xbb, (int)(XELEM / 8));

    route_topk  <<<T_TOK / 256, 256, 0, stream>>>(logits, tidx, tw);
    build_lists <<<NEXP, T_TOK, 0, stream>>>(tidx, cnt, lst);
    moe_expert_tile<<<dim3(T_TOK / MTILE, NEXP), 256, 0, stream>>>(
        xbb, gwb, uwb, dwb, cnt, lst, tw, partial);
    combine_topk<<<(T_TOK * HDIM / 4) / 256, 256, 0, stream>>>(partial, (float*)d_out);
}
